// AttentionBasisSynthesizer_91044716741303
// MI455X (gfx1250) — matrix operand order repaired and checked by reading, not compiled, not run on hardware
//
#include <hip/hip_runtime.h>
#include <hip/hip_bf16.h>
#include <cmath>

// Problem constants from the reference
#define B_       8
#define S_       2048
#define D_       128
#define N_ATOMS  64
#define MAX_LEN  4096
#define P_       16

// Tiling
#define BLOCK_Q  128   // per workgroup (8 waves x 16 rows)
#define BLOCK_K  64    // keys per iteration (4 x 16-key WMMA column tiles)
#define NWAVES   8
#define KSTRIDE  136   // 128 + pad (f16) to break LDS bank alignment
#define PSTRIDE  72    // 64 + pad

#define LOG2E    1.4426950408889634f

typedef __attribute__((ext_vector_type(16))) _Float16 v16h;
typedef __attribute__((ext_vector_type(8)))  float    v8f;
typedef int vint4 __attribute__((vector_size(16)));

// Async global->LDS staging (gfx1250). Fallback: plain f16 b128 copies.
#if __has_builtin(__builtin_amdgcn_global_load_async_to_lds_b128) && \
    __has_builtin(__builtin_amdgcn_s_wait_asynccnt)
#define USE_ASYNC 1
#define B128P(p) ((vint4*)(void*)(p))
#else
#define USE_ASYNC 0
#endif

// DPP16 controls (stay within each 16-lane row; wave32 = 2 rows)
#define DPP_QUAD_XOR1   0xB1   // quad_perm [1,0,3,2]
#define DPP_QUAD_XOR2   0x4E   // quad_perm [2,3,0,1]
#define DPP_HALF_MIRROR 0x141  // mirror within groups of 8
#define DPP_ROW_MIRROR  0x140  // mirror within row of 16

template <int CTRL>
__device__ __forceinline__ float dpp_maxf(float x) {
  int yi = __builtin_amdgcn_update_dpp(0, __float_as_int(x), CTRL, 0xF, 0xF, true);
  return fmaxf(x, __int_as_float(yi));
}

// ---------------------------------------------------------------------------
// Kernel 1: bias[k] = window[k] * sum_p gains[p] * W[atom[p], (k - shift[p]) mod S]
// ---------------------------------------------------------------------------
__global__ __launch_bounds__(256)
void bias_kernel(const float* __restrict__ waveforms, const float* __restrict__ gains,
                 const float* __restrict__ window, const int* __restrict__ atom_idx,
                 const int* __restrict__ shifts, float* __restrict__ bias) {
  int k = blockIdx.x * blockDim.x + threadIdx.x;
  if (k >= S_) return;
  float acc = 0.0f;
#pragma unroll
  for (int p = 0; p < P_; ++p) {
    int src = k - shifts[p];           // shifts in [0,S)
    if (src < 0) src += S_;
    acc += gains[p] * waveforms[(size_t)atom_idx[p] * MAX_LEN + src];
  }
  bias[k] = acc * window[k];
}

// ---------------------------------------------------------------------------
// Kernel 2: pack K and V to f16 once (removes all cvt from the hot loop)
// ---------------------------------------------------------------------------
__global__ __launch_bounds__(256)
void pack_kernel(const float* __restrict__ K, const float* __restrict__ V,
                 _Float16* __restrict__ Kh, _Float16* __restrict__ Vh) {
  size_t i = (size_t)blockIdx.x * blockDim.x + threadIdx.x;
  if (i >= (size_t)B_ * S_ * D_) return;
  Kh[i] = (_Float16)K[i];
  Vh[i] = (_Float16)V[i];
}

// ---------------------------------------------------------------------------
// Kernel 3: flash attention, out = softmax(Q K^T + bias[k]) V  via WMMA f16,
// with double-buffered async global->LDS tile staging.
// ---------------------------------------------------------------------------
__global__ __launch_bounds__(256)
void fa_kernel(const float* __restrict__ Q, const _Float16* __restrict__ Kh,
               const _Float16* __restrict__ Vh, const float* __restrict__ bias,
               float* __restrict__ Out) {
  __shared__ _Float16 Ksh[2][BLOCK_K][KSTRIDE];
  __shared__ _Float16 Vsh[2][BLOCK_K][KSTRIDE];
  __shared__ _Float16 Psh[NWAVES][16][PSTRIDE];
  __shared__ float    Bsh[BLOCK_K];

  const int tid  = threadIdx.x;
  const int wave = tid >> 5;
  const int lane = tid & 31;
  const int half = lane >> 4;   // 0 for lanes 0-15, 1 for lanes 16-31
  const int ln   = lane & 15;

  const int b  = blockIdx.y;
  const int qb = blockIdx.x * BLOCK_Q + wave * 16;

  // staging geometry: thread owns 64 contiguous bytes of one row per array
  const int srow = tid >> 2;            // 0..63
  const int scol = (tid & 3) * 32;      // f16 index: 0,32,64,96

  // --- Load Q fragment in WMMA A-layout: lane = row (qb+ln), 4 chunks of K=32.
  const float* qrow = Q + ((size_t)b * S_ + qb + ln) * D_;
  v16h qa[4];
#pragma unroll
  for (int c = 0; c < 4; ++c) {
    const int base = 32 * c + 16 * half;
#pragma unroll
    for (int e = 0; e < 8; ++e) {
      qa[c][e]     = (_Float16)qrow[base + e];
      qa[c][8 + e] = (_Float16)qrow[base + 8 + e];
    }
  }

  // All-ones B fragment: osum = P x Ones accumulates the softmax denominator.
  v16h ones;
#pragma unroll
  for (int e = 0; e < 16; ++e) ones[e] = (_Float16)1.0f;

  v8f o[8] = {};
  v8f osum = {};
  float mrow[8];
#pragma unroll
  for (int r = 0; r < 8; ++r) mrow[r] = -3.0e38f;

  // --- Preload tile 0 into buffer 0 (8 x b128 per thread: 4 K + 4 V)
  {
    const _Float16* kg = Kh + ((size_t)b * S_ + srow) * D_ + scol;
    const _Float16* vg = Vh + ((size_t)b * S_ + srow) * D_ + scol;
#pragma unroll
    for (int i = 0; i < 4; ++i) {
#if USE_ASYNC
      __builtin_amdgcn_global_load_async_to_lds_b128(B128P(kg + 8 * i), B128P(&Ksh[0][srow][scol + 8 * i]), 0, 0);
      __builtin_amdgcn_global_load_async_to_lds_b128(B128P(vg + 8 * i), B128P(&Vsh[0][srow][scol + 8 * i]), 0, 0);
#else
      *(float4*)&Ksh[0][srow][scol + 8 * i] = *(const float4*)(kg + 8 * i);
      *(float4*)&Vsh[0][srow][scol + 8 * i] = *(const float4*)(vg + 8 * i);
#endif
    }
  }

  const int nIter = S_ / BLOCK_K;
  for (int it = 0; it < nIter; ++it) {
    const int kt  = it * BLOCK_K;
    const int buf = it & 1;

    __syncthreads();   // all waves done computing from buffer buf^1

    if (tid < BLOCK_K) Bsh[tid] = bias[kt + tid] * LOG2E;

    // Issue next tile into the other buffer while we compute this one
    if (it + 1 < nIter) {
      const _Float16* kg = Kh + ((size_t)b * S_ + kt + BLOCK_K + srow) * D_ + scol;
      const _Float16* vg = Vh + ((size_t)b * S_ + kt + BLOCK_K + srow) * D_ + scol;
#pragma unroll
      for (int i = 0; i < 4; ++i) {
#if USE_ASYNC
        __builtin_amdgcn_global_load_async_to_lds_b128(B128P(kg + 8 * i), B128P(&Ksh[buf ^ 1][srow][scol + 8 * i]), 0, 0);
        __builtin_amdgcn_global_load_async_to_lds_b128(B128P(vg + 8 * i), B128P(&Vsh[buf ^ 1][srow][scol + 8 * i]), 0, 0);
#else
        *(float4*)&Ksh[buf ^ 1][srow][scol + 8 * i] = *(const float4*)(kg + 8 * i);
        *(float4*)&Vsh[buf ^ 1][srow][scol + 8 * i] = *(const float4*)(vg + 8 * i);
#endif
      }
#if USE_ASYNC
      __builtin_amdgcn_s_wait_asynccnt(8);  // retire tile `it` (in-order), keep tile it+1 in flight
#endif
    } else {
#if USE_ASYNC
      __builtin_amdgcn_s_wait_asynccnt(0);
#endif
    }
    __syncthreads();   // tile `it` visible to all waves

    // --- S = Q K^T for 4 interleaved 16-key column tiles (tile j -> key kt+4*ln+j)
    v8f s[4] = {};
#pragma unroll
    for (int c = 0; c < 4; ++c) {
      const int base = 32 * c + 16 * half;
#pragma unroll
      for (int j = 0; j < 4; ++j) {
        v16h kb;
#pragma unroll
        for (int e = 0; e < 16; ++e)
          kb[e] = Ksh[buf][4 * ln + j][base + e];
        s[j] = __builtin_amdgcn_wmma_f32_16x16x32_f16(false, qa[c], false, kb,
                                                      (short)0, s[j], false, false);
      }
    }

    float bj[4];
#pragma unroll
    for (int j = 0; j < 4; ++j) bj[j] = Bsh[4 * ln + j];

    // --- Online softmax in log2 domain
    float t[8];
#pragma unroll
    for (int r = 0; r < 8; ++r) {
#pragma unroll
      for (int j = 0; j < 4; ++j)
        s[j][r] = fmaf(s[j][r], LOG2E, bj[j]);
      t[r] = fmaxf(fmaxf(s[0][r], s[1][r]), fmaxf(s[2][r], s[3][r]));
    }
#pragma unroll
    for (int r = 0; r < 8; ++r) t[r] = dpp_maxf<DPP_QUAD_XOR1>(t[r]);
#pragma unroll
    for (int r = 0; r < 8; ++r) t[r] = dpp_maxf<DPP_QUAD_XOR2>(t[r]);
#pragma unroll
    for (int r = 0; r < 8; ++r) t[r] = dpp_maxf<DPP_HALF_MIRROR>(t[r]);
#pragma unroll
    for (int r = 0; r < 8; ++r) t[r] = dpp_maxf<DPP_ROW_MIRROR>(t[r]);

#pragma unroll
    for (int r = 0; r < 8; ++r) {
      const float mnew  = fmaxf(mrow[r], t[r]);
      const float alpha = exp2f(mrow[r] - mnew);
      mrow[r] = mnew;
      float p[4];
#pragma unroll
      for (int j = 0; j < 4; ++j) p[j] = exp2f(s[j][r] - mnew);
      osum[r] *= alpha;
#pragma unroll
      for (int dc = 0; dc < 8; ++dc) o[dc][r] *= alpha;
      union { _Float16 h[4]; unsigned long long u; } pk;
      pk.h[0] = (_Float16)p[0];
      pk.h[1] = (_Float16)p[1];
      pk.h[2] = (_Float16)p[2];
      pk.h[3] = (_Float16)p[3];
      *(unsigned long long*)&Psh[wave][r + 8 * half][4 * ln] = pk.u;
    }
    __syncthreads();

    // --- O += P V ; osum += P * Ones
    v16h pa0, pa1;
#pragma unroll
    for (int e = 0; e < 8; ++e) {
      pa0[e]     = Psh[wave][ln][8 * half + e];
      pa0[8 + e] = Psh[wave][ln][16 + 8 * half + e];
      pa1[e]     = Psh[wave][ln][32 + 8 * half + e];
      pa1[8 + e] = Psh[wave][ln][48 + 8 * half + e];
    }
    osum = __builtin_amdgcn_wmma_f32_16x16x32_f16(false, pa0, false, ones,
                                                  (short)0, osum, false, false);
    osum = __builtin_amdgcn_wmma_f32_16x16x32_f16(false, pa1, false, ones,
                                                  (short)0, osum, false, false);
#pragma unroll
    for (int dc = 0; dc < 8; ++dc) {
      v16h vb0, vb1;
#pragma unroll
      for (int e = 0; e < 16; ++e) {
        vb0[e] = Vsh[buf][8 * half + e + (e & 8)][dc * 16 + ln];
        vb1[e] = Vsh[buf][32 + 8 * half + e + (e & 8)][dc * 16 + ln];
      }
      o[dc] = __builtin_amdgcn_wmma_f32_16x16x32_f16(false, pa0, false, vb0,
                                                     (short)0, o[dc], false, false);
      o[dc] = __builtin_amdgcn_wmma_f32_16x16x32_f16(false, pa1, false, vb1,
                                                     (short)0, o[dc], false, false);
    }
  }

  // --- Normalize by row sum and store (fp32 output)
#pragma unroll
  for (int r = 0; r < 8; ++r) {
    const float inv = 1.0f / osum[r];
    float* orow = Out + ((size_t)b * S_ + qb + r + 8 * half) * D_;
#pragma unroll
    for (int dc = 0; dc < 8; ++dc)
      orow[dc * 16 + ln] = o[dc][r] * inv;
  }
}

// ---------------------------------------------------------------------------
extern "C" void kernel_launch(void* const* d_in, const int* in_sizes, int n_in,
                              void* d_out, int out_size, void* d_ws, size_t ws_size,
                              hipStream_t stream) {
  (void)in_sizes; (void)n_in; (void)out_size; (void)ws_size;
  const float* Qp        = (const float*)d_in[0];
  const float* Kp        = (const float*)d_in[1];
  const float* Vp        = (const float*)d_in[2];
  const float* waveforms = (const float*)d_in[3];
  const float* gains     = (const float*)d_in[4];
  const float* window    = (const float*)d_in[5];
  const int*   atom_idx  = (const int*)d_in[6];
  const int*   shifts    = (const int*)d_in[7];

  // Workspace layout: [bias: S_ f32][Kh: B*S*D f16][Vh: B*S*D f16]  (~8.4 MB)
  float*    bias = (float*)d_ws;
  _Float16* Kh   = (_Float16*)((char*)d_ws + 16384);
  _Float16* Vh   = Kh + (size_t)B_ * S_ * D_;
  float*    out  = (float*)d_out;

  bias_kernel<<<dim3((S_ + 255) / 256), dim3(256), 0, stream>>>(
      waveforms, gains, window, atom_idx, shifts, bias);

  const size_t nelt = (size_t)B_ * S_ * D_;
  pack_kernel<<<dim3((unsigned)((nelt + 255) / 256)), dim3(256), 0, stream>>>(
      Kp, Vp, Kh, Vh);

  fa_kernel<<<dim3(S_ / BLOCK_Q, B_), dim3(256), 0, stream>>>(
      Qp, Kh, Vh, bias, out);
}
